// NeuralHawkes_5935644803469
// MI455X (gfx1250) — matrix operand order repaired and checked by reading, not compiled, not run on hardware
//
#include <hip/hip_runtime.h>

// Problem sizes (fixed by reference)
#define N_EV   8192
#define DIN    512
#define DH     1024
#define NGATE  5
#define RBLOCKS 40          // 40 blocks * 8 waves = 320 waves = 5 gates * 64 tiles
#define RTHREADS 256

typedef __bf16 bf16;
typedef __attribute__((ext_vector_type(16))) __bf16 v16bf;
typedef __attribute__((ext_vector_type(8)))  __bf16 v8bf;
typedef __attribute__((ext_vector_type(8)))  float  v8f;
typedef __attribute__((ext_vector_type(4)))  unsigned int v4u;
typedef __attribute__((ext_vector_type(8)))  int v8i_t;
typedef __attribute__((ext_vector_type(4)))  int v4i_t;

__device__ inline bf16 f2bf(float f) {
  union { float f; unsigned u; } v; v.f = f;
  unsigned r = v.u + 0x7FFFu + ((v.u >> 16) & 1u);   // round-to-nearest-even
  unsigned short h = (unsigned short)(r >> 16);
  bf16 out; __builtin_memcpy(&out, &h, 2); return out;
}

// ---------------------------------------------------------------- fp32 -> bf16
__global__ void nh_cvt_bf16(const float* __restrict__ in, bf16* __restrict__ out, int n) {
  for (int i = blockIdx.x * blockDim.x + threadIdx.x; i < n; i += gridDim.x * blockDim.x)
    out[i] = f2bf(in[i]);
}

// ---------------------------------------------------------------- WMMA GEMM
// Out[M,Nout] = A[M,K](bf16) @ W[Nout,K](bf16)^T + bias.  Row-major everywhere.
// Wave tile: 16(M) x 64(N), 4 accumulators, K-loop step 32.
__global__ void __launch_bounds__(256)
nh_gemm_bf16(const bf16* __restrict__ A, const bf16* __restrict__ W,
             const float* __restrict__ bias,
             float* __restrict__ outF, bf16* __restrict__ outB,
             int M, int K, int Nout) {
  const int lane = threadIdx.x & 31;
  const int wid  = threadIdx.x >> 5;
  const int lo   = lane & 15;
  const int hi   = lane >> 4;
  const int ntiles = Nout >> 6;
  const int gw = blockIdx.x * (blockDim.x >> 5) + wid;
  if (gw >= (M >> 4) * ntiles) return;
  const int row0 = (gw / ntiles) << 4;
  const int col0 = (gw % ntiles) << 6;

  v8f acc0 = {}, acc1 = {}, acc2 = {}, acc3 = {};

  // A frag per-lane base: row (row0+lo), K offset hi*16 (and +8 for upper halves)
  const bf16* arow = A + (size_t)(row0 + lo) * K + hi * 16;
  // B frag per-lane base: weight row (col0 + j*16 + lo), K offset hi*16
  const bf16* w0 = W + (size_t)(col0 +  0 + lo) * K + hi * 16;
  const bf16* w1 = W + (size_t)(col0 + 16 + lo) * K + hi * 16;
  const bf16* w2 = W + (size_t)(col0 + 32 + lo) * K + hi * 16;
  const bf16* w3 = W + (size_t)(col0 + 48 + lo) * K + hi * 16;

  for (int k0 = 0; k0 < K; k0 += 32) {
    __builtin_prefetch(arow + k0 + 128, 0, 0);  // global_prefetch_b8
    v8bf a0 = *(const v8bf*)(arow + k0);
    v8bf a1 = *(const v8bf*)(arow + k0 + 8);
    v16bf a;
#pragma unroll
    for (int i = 0; i < 8; ++i) { a[i] = a0[i]; a[8 + i] = a1[i]; }
    v16bf b0 = *(const v16bf*)(w0 + k0);
    v16bf b1 = *(const v16bf*)(w1 + k0);
    v16bf b2 = *(const v16bf*)(w2 + k0);
    v16bf b3 = *(const v16bf*)(w3 + k0);
    acc0 = __builtin_amdgcn_wmma_f32_16x16x32_bf16(false, a, false, b0, (short)0, acc0, false, false);
    acc1 = __builtin_amdgcn_wmma_f32_16x16x32_bf16(false, a, false, b1, (short)0, acc1, false, false);
    acc2 = __builtin_amdgcn_wmma_f32_16x16x32_bf16(false, a, false, b2, (short)0, acc2, false, false);
    acc3 = __builtin_amdgcn_wmma_f32_16x16x32_bf16(false, a, false, b3, (short)0, acc3, false, false);
  }

#pragma unroll
  for (int r = 0; r < 8; ++r) {
    const size_t m = (size_t)(row0 + r + hi * 8);
    const int c0 = col0 + lo;
    float v0 = acc0[r] + bias[c0];
    float v1 = acc1[r] + bias[c0 + 16];
    float v2 = acc2[r] + bias[c0 + 32];
    float v3 = acc3[r] + bias[c0 + 48];
    if (outF) {
      outF[m * Nout + c0]      = v0;
      outF[m * Nout + c0 + 16] = v1;
      outF[m * Nout + c0 + 32] = v2;
      outF[m * Nout + c0 + 48] = v3;
    }
    if (outB) {
      outB[m * Nout + c0]      = f2bf(v0);
      outB[m * Nout + c0 + 16] = f2bf(v1);
      outB[m * Nout + c0 + 32] = f2bf(v2);
      outB[m * Nout + c0 + 48] = f2bf(v3);
    }
  }
}

// ---------------------------------------------------------------- init state
__global__ void nh_init(float* c, float* cb, float* pdec, bf16* hbuf, int* bar) {
  int i = blockIdx.x * blockDim.x + threadIdx.x;
  if (i < DH) { c[i] = 0.f; cb[i] = 0.f; pdec[i] = 0.f; hbuf[i] = f2bf(0.f); }
  if (i == 0) *bar = 0;
}

// ---------------------------------------------------------------- grid barrier
__device__ inline void gridbar(int* bar, int phase) {
  __threadfence();
  __syncthreads();
  if (threadIdx.x == 0) {
    atomicAdd(bar, 1);
    const int target = RBLOCKS * (phase + 1);   // monotonic counter, never reset
    while (__hip_atomic_load(bar, __ATOMIC_RELAXED, __HIP_MEMORY_SCOPE_AGENT) < target)
      __builtin_amdgcn_s_sleep(1);
    __threadfence();
  }
  __syncthreads();
}

// ---------------------------------------------------------------- recurrence
// Persistent kernel. 320 waves: wave (g,n) owns gate g, output cols [16n,16n+16).
// Per step: TDM pulls h (2KB bf16) from global into LDS (TENSORcnt-tracked),
// then each wave runs the WMMA GEMV with A-fragments fed from LDS, then 1024
// threads do the elementwise CT-LSTM update between two grid barriers.
__global__ void __launch_bounds__(RTHREADS, 1)
nh_recur(const float* __restrict__ times,
         const float* __restrict__ gxi, const float* __restrict__ gxf,
         const float* __restrict__ gxo, const float* __restrict__ gxz,
         const float* __restrict__ gxd,
         const bf16* __restrict__ U,            // [5][DH][DH] bf16, row-major
         float* __restrict__ Hout, float* __restrict__ Cout,
         float* __restrict__ Decout, float* __restrict__ cbfin,
         float* __restrict__ c, float* __restrict__ cb, float* __restrict__ pdec,
         bf16* __restrict__ hbuf, float* __restrict__ gbuf,
         int* bar, int nsteps) {
  __shared__ __align__(16) bf16 sh_h[DH];      // current h vector, staged by TDM

  const int lane = threadIdx.x & 31;
  const int wid  = threadIdx.x >> 5;
  const int lo   = lane & 15;
  const int hi   = lane >> 4;
  const int gw   = blockIdx.x * (RTHREADS / 32) + wid;   // 0..319
  const int g    = gw >> 6;                              // gate 0..4
  const int h0   = (gw & 63) << 4;                       // output col base
  const int gtid = blockIdx.x * RTHREADS + threadIdx.x;

  const bf16* Ug  = U + ((size_t)g * DH + (h0 + lo)) * DH + hi * 16;
  const bf16* hb0 = sh_h + hi * 16;

#if __has_builtin(__builtin_amdgcn_tensor_load_to_lds)
  // Tensor DMA descriptor (D#) for a 1024x1 bf16 tile: global hbuf -> LDS sh_h.
  const unsigned long long ga = (unsigned long long)(const void*)hbuf;
  const unsigned lds_off = (unsigned)(size_t)(void*)sh_h;
  v4u g0;
  g0[0] = 1u;                                            // count=1, user mode
  g0[1] = lds_off;                                       // lds_addr
  g0[2] = (unsigned)ga;                                  // global_addr[31:0]
  g0[3] = (unsigned)((ga >> 32) & 0x01FFFFFFull) | (2u << 30);  // addr[56:32], type=2
  v8i_t g1;
  g1[0] = (int)(1u << 16);                               // data_size=1 (2 bytes)
  g1[1] = (int)(((unsigned)DH & 0xFFFFu) << 16);         // tensor_dim0 lo16 = 1024
  g1[2] = (int)(((unsigned)DH >> 16) | (1u << 16));      // tensor_dim0 hi | tensor_dim1=1
  g1[3] = (int)((unsigned)DH << 16);                     // tile_dim0 = 1024
  g1[4] = 1;                                             // tile_dim1 = 1
  g1[5] = DH;                                            // tensor_dim0_stride = 1024
  g1[6] = 0; g1[7] = 0;
  const v4i_t gz = {0, 0, 0, 0};
#if defined(__clang_major__) && __clang_major__ >= 23
  const v8i_t gz8 = {0, 0, 0, 0, 0, 0, 0, 0};
#endif
#endif

  for (int s = 0; s < nsteps; ++s) {
    // ---- stage h into LDS (Tensor Data Mover; TENSORcnt-tracked)
#if __has_builtin(__builtin_amdgcn_tensor_load_to_lds)
    if (threadIdx.x == 0) {
#if defined(__clang_major__) && __clang_major__ >= 23
      __builtin_amdgcn_tensor_load_to_lds(g0, g1, gz, gz, gz8, 0);
#else
      __builtin_amdgcn_tensor_load_to_lds(g0, g1, gz, gz, 0);
#endif
      __builtin_amdgcn_s_wait_tensorcnt(0);
    }
#else
    for (int i = threadIdx.x; i < DH; i += RTHREADS) sh_h[i] = hbuf[i];
#endif
    __syncthreads();

    // ---- phase A: gsum[g, h0..h0+15] = h @ Ug^T (WMMA, row 0 of D is exact)
    v8f acc = {};
    for (int k0 = 0; k0 < DH; k0 += 32) {
      v8bf a0 = *(const v8bf*)(hb0 + k0);        // ds_load: broadcast h chunk to all rows
      v8bf a1 = *(const v8bf*)(hb0 + k0 + 8);
      v16bf a;
#pragma unroll
      for (int i = 0; i < 8; ++i) { a[i] = a0[i]; a[8 + i] = a1[i]; }
      v16bf b = *(const v16bf*)(Ug + k0);
      acc = __builtin_amdgcn_wmma_f32_16x16x32_bf16(false, a, false, b, (short)0, acc, false, false);
    }
    if (lane < 16) gbuf[g * DH + h0 + lane] = acc[0];   // D row M=0: VGPR0, lanes 0-15
    gridbar(bar, 2 * s);

    // ---- phase B: elementwise CT-LSTM update (1024 owner threads)
    if (gtid < DH) {
      const float dt = (s == 0) ? 0.0f : (times[s] - times[s - 1]);
      float cv = c[gtid], cbv = cb[gtid], pd = pdec[gtid];
      cv = cbv + (cv - cbv) * __expf(-pd * dt);
      const size_t ro = (size_t)s * DH + gtid;
      const float ai = gxi[ro] + gbuf[0 * DH + gtid];
      const float af = gxf[ro] + gbuf[1 * DH + gtid];
      const float ao = gxo[ro] + gbuf[2 * DH + gtid];
      const float az = gxz[ro] + gbuf[3 * DH + gtid];
      const float ad = gxd[ro] + gbuf[4 * DH + gtid];
      const float ig = 1.0f / (1.0f + __expf(-ai));
      const float fg = 1.0f / (1.0f + __expf(-af));
      const float og = 1.0f / (1.0f + __expf(-ao));
      const float z  = tanhf(az);
      const float cn = fg * cv + ig * z;
      cbv = fg * cbv + ig * z;
      const float dec = (ad > 20.0f) ? ad : log1pf(__expf(ad));
      const float hn = og * tanhf(cn);
      c[gtid] = cn; cb[gtid] = cbv; pdec[gtid] = dec;
      hbuf[gtid] = f2bf(hn);
      Hout[ro] = hn; Cout[ro] = cn; Decout[ro] = dec;
      if (s == nsteps - 1) cbfin[gtid] = cbv;
    }
    gridbar(bar, 2 * s + 1);
  }
}

// ---------------------------------------------------------------- launcher
extern "C" void kernel_launch(void* const* d_in, const int* in_sizes, int n_in,
                              void* d_out, int out_size, void* d_ws, size_t ws_size,
                              hipStream_t stream) {
  const float* times = (const float*)d_in[0];
  const float* marks = (const float*)d_in[1];
  // d_in[2] = T (unused)
  const float* Wp = (const float*)d_in[3];
  const float* bp = (const float*)d_in[4];
  const float* Wg[NGATE] = { (const float*)d_in[5],  (const float*)d_in[7],
                             (const float*)d_in[9],  (const float*)d_in[11],
                             (const float*)d_in[13] };
  const float* bg[NGATE] = { (const float*)d_in[6],  (const float*)d_in[8],
                             (const float*)d_in[10], (const float*)d_in[12],
                             (const float*)d_in[14] };
  const float* Uin[NGATE] = { (const float*)d_in[15], (const float*)d_in[16],
                              (const float*)d_in[17], (const float*)d_in[18],
                              (const float*)d_in[19] };

  char* wp = (char*)d_ws;
  auto take = [&](size_t bytes) -> char* {
    char* p = wp; wp += (bytes + 255) & ~(size_t)255; return p;
  };
  bf16*  marksb = (bf16*)take((size_t)N_EV * DIN * 2);
  bf16*  Xb     = (bf16*)take((size_t)N_EV * DH * 2);
  bf16*  Wpb    = (bf16*)take((size_t)DH * DIN * 2);
  bf16*  Wgb    = (bf16*)take((size_t)NGATE * DH * DH * 2);
  bf16*  Ub     = (bf16*)take((size_t)NGATE * DH * DH * 2);
  float* gx     = (float*)take((size_t)NGATE * N_EV * DH * 4);
  float* cst    = (float*)take(DH * 4);
  float* cbst   = (float*)take(DH * 4);
  float* pdec   = (float*)take(DH * 4);
  bf16*  hbuf   = (bf16*)take(DH * 2);
  float* gbuf   = (float*)take((size_t)NGATE * DH * 4);
  int*   bar    = (int*)take(256);

  auto cvt = [&](const float* src, bf16* dst, int n) {
    int blocks = (n + 255) / 256; if (blocks > 4096) blocks = 4096;
    nh_cvt_bf16<<<blocks, 256, 0, stream>>>(src, dst, n);
  };
  cvt(marks, marksb, N_EV * DIN);
  cvt(Wp, Wpb, DH * DIN);
  for (int g = 0; g < NGATE; ++g) {
    cvt(Wg[g],  Wgb + (size_t)g * DH * DH, DH * DH);
    cvt(Uin[g], Ub  + (size_t)g * DH * DH, DH * DH);
  }

  // X = marks @ Wp^T + bp  (bf16 out, feeds the 5 gate GEMMs)
  {
    int tiles = (N_EV / 16) * (DH / 64);
    nh_gemm_bf16<<<tiles / 8, 256, 0, stream>>>(marksb, Wpb, bp,
                                                nullptr, Xb, N_EV, DIN, DH);
  }
  // gx_g = X @ Wg^T + bg  (fp32 out)
  for (int g = 0; g < NGATE; ++g) {
    int tiles = (N_EV / 16) * (DH / 64);
    nh_gemm_bf16<<<tiles / 8, 256, 0, stream>>>(Xb, Wgb + (size_t)g * DH * DH, bg[g],
                                                gx + (size_t)g * N_EV * DH, nullptr,
                                                N_EV, DH, DH);
  }

  float* out  = (float*)d_out;
  float* Hout = out;
  float* Cout = out + (size_t)N_EV * DH;
  float* Dout = out + (size_t)2 * N_EV * DH;
  float* cbf  = out + (size_t)3 * N_EV * DH;

  nh_init<<<4, 256, 0, stream>>>(cst, cbst, pdec, hbuf, bar);
  nh_recur<<<RBLOCKS, RTHREADS, 0, stream>>>(
      times,
      gx + (size_t)0 * N_EV * DH, gx + (size_t)1 * N_EV * DH,
      gx + (size_t)2 * N_EV * DH, gx + (size_t)3 * N_EV * DH,
      gx + (size_t)4 * N_EV * DH,
      Ub, Hout, Cout, Dout, cbf,
      cst, cbst, pdec, hbuf, gbuf, bar, N_EV);
}
